// VectorQuantizer_34248069218960
// MI455X (gfx1250) — matrix operand order repaired and checked by reading, not compiled, not run on hardware
//
#include <hip/hip_runtime.h>
#include <math.h>

typedef __attribute__((ext_vector_type(8)))  _Float16 h8;
typedef __attribute__((ext_vector_type(16))) _Float16 v16h;
typedef __attribute__((ext_vector_type(8)))  float    v8f;

#define D_DIM   256
#define K_CODES 1024
#define HW      4096
#define N_PTS   65536
#define TOTAL_ELEMS 16777216
#define M_TILE  64    // points per block (4 x 16-row WMMA tiles)
#define LDA     272   // padded LDS row (f16 elems) -> 544B stride, rotates banks

typedef unsigned long long u64;

// Order-preserving f32 -> u32 map (ascending float order == ascending uint)
__device__ __forceinline__ unsigned sortable(float f) {
    unsigned u = __float_as_uint(f);
    return u ^ ((unsigned)((int)u >> 31) | 0x80000000u);
}

// ---------------------------------------------------------------------------
// Prep: E [256][1024] f32 -> Et [1024][256] f16 (row-contiguous code vectors),
// enorm[k] = sum_d E[d][k]^2 (f32), and zero the histogram counts.
// ---------------------------------------------------------------------------
__global__ void vq_prep(const float* __restrict__ E, _Float16* __restrict__ Et,
                        float* __restrict__ enorm, unsigned* __restrict__ counts) {
    const int k = blockIdx.x;
    const int d = threadIdx.x;
    float v = E[(size_t)d * K_CODES + k];
    Et[(size_t)k * D_DIM + d] = (_Float16)v;
    __shared__ float red[256];
    red[d] = v * v;
    __syncthreads();
    for (int s = 128; s > 0; s >>= 1) {
        if (d < s) red[d] += red[d + s];
        __syncthreads();
    }
    if (d == 0) { enorm[k] = red[0]; counts[k] = 0u; }
}

// ---------------------------------------------------------------------------
// Main fused kernel: per block, a 64-point tile vs all 1024 codes.
//   score = ||e||^2 - 2 * <x, e>   (||x||^2 constant per row -> dropped)
// Argmin kept as branchless min over (sortable(score) << 32 | code) keys.
// ---------------------------------------------------------------------------
__global__ __launch_bounds__(256) void vq_main(
    const float* __restrict__ x, const float* __restrict__ E,
    const _Float16* __restrict__ Et, const float* __restrict__ enorm,
    float* __restrict__ out_q, float* __restrict__ out_idx,
    unsigned* __restrict__ counts, float* __restrict__ partials) {

    __shared__ __align__(16) _Float16 As[M_TILE * LDA];  // f16 tile for WMMA A
    __shared__ float Af[M_TILE * LDA];                   // f32 tile for SSE
    __shared__ u64   red_kv[8][M_TILE];
    __shared__ int   bestk_sh[M_TILE];
    __shared__ float sse_red[256];

    const int tid  = threadIdx.x;
    const int lane = tid & 31;
    const int wave = tid >> 5;
    const int n0   = blockIdx.x * M_TILE;        // first point of tile
    const int b    = n0 >> 12;                   // n / HW  (HW = 4096)
    const int hw0  = n0 & (HW - 1);
    const float* xb = x + (size_t)b * D_DIM * HW + hw0;

    // Stage x tile into LDS with transpose: x[b][d][hw] -> tile[r][d].
    {
        const int r  = tid & 63;
        const int dg = tid >> 6;                 // 0..3
        for (int p = 0; p < 64; ++p) {
            const int d = dg + p * 4;
            float v = xb[(size_t)d * HW + r];
            Af[r * LDA + d] = v;
            As[r * LDA + d] = (_Float16)v;
        }
    }
    __syncthreads();

    const int row  = lane & 15;
    const int half = lane >> 4;
    const int col  = lane & 15;

    u64 best[4][8];
#pragma unroll
    for (int mt = 0; mt < 4; ++mt)
#pragma unroll
        for (int j = 0; j < 8; ++j) best[mt][j] = ~0ull;

#pragma unroll 1
    for (int it = 0; it < 8; ++it) {
        const int code = it * 128 + wave * 16 + col;
        // B layout: lane half h holds contiguous K = 16h..16h+15 of column col
        const _Float16* bp = Et + (size_t)code * D_DIM + half * 16;
        v16h bfrag[8];
#pragma unroll
        for (int kd = 0; kd < 8; ++kd) {
            h8 blo = *(const h8*)(bp + kd * 32);
            h8 bhi = *(const h8*)(bp + kd * 32 + 8);
            bfrag[kd] = __builtin_shufflevector(blo, bhi,
                0, 1, 2, 3, 4, 5, 6, 7, 8, 9, 10, 11, 12, 13, 14, 15);
        }
        const float en = enorm[code];

#pragma unroll
        for (int mt = 0; mt < 4; ++mt) {
            const int arow = mt * 16 + row;
            v8f acc = {};
#pragma unroll
            for (int kd = 0; kd < 8; ++kd) {
                // A layout: lane half h holds contiguous K = base..base+15
                const int base = kd * 32 + half * 16;
                h8 alo = *(const h8*)&As[arow * LDA + base];
                h8 ahi = *(const h8*)&As[arow * LDA + base + 8];
                v16h a = __builtin_shufflevector(alo, ahi,
                    0, 1, 2, 3, 4, 5, 6, 7, 8, 9, 10, 11, 12, 13, 14, 15);
                acc = __builtin_amdgcn_wmma_f32_16x16x32_f16(
                    false, a, false, bfrag[kd], (short)0, acc, false, false);
            }
#pragma unroll
            for (int j = 0; j < 8; ++j) {
                const float s = fmaf(-2.0f, acc[j], en);
                const u64 key = ((u64)sortable(s) << 32) | (unsigned)code;
                best[mt][j] = key < best[mt][j] ? key : best[mt][j];
            }
        }
    }

    // Argmin across the 16 lanes of each half (same rows, different codes).
#pragma unroll
    for (int off = 1; off < 16; off <<= 1) {
#pragma unroll
        for (int mt = 0; mt < 4; ++mt)
#pragma unroll
            for (int j = 0; j < 8; ++j) {
                const u64 o = __shfl_xor(best[mt][j], off, 32);
                best[mt][j] = o < best[mt][j] ? o : best[mt][j];
            }
    }
    if ((lane & 15) == 0) {
#pragma unroll
        for (int mt = 0; mt < 4; ++mt)
#pragma unroll
            for (int j = 0; j < 8; ++j)
                red_kv[wave][mt * 16 + half * 8 + j] = best[mt][j];
    }
    __syncthreads();

    // Cross-wave argmin + emit indices + histogram.
    if (tid < M_TILE) {
        u64 kv = red_kv[0][tid];
        for (int w = 1; w < 8; ++w) {
            const u64 o = red_kv[w][tid];
            kv = o < kv ? o : kv;
        }
        const int k = (int)(kv & 0xFFFFFFFFu);
        bestk_sh[tid] = k;
        out_idx[n0 + tid] = (float)k;
        atomicAdd(&counts[k], 1u);
    }
    __syncthreads();

    // Gather codes (f32), write quantized (== quantized_ste fwd), accum SSE.
    float sse = 0.0f;
    {
        const int r  = tid & 63;
        const int dg = tid >> 6;
        const int k  = bestk_sh[r];
        float* ob = out_q + (size_t)b * D_DIM * HW + hw0;
        for (int p = 0; p < 64; ++p) {
            const int d = dg + p * 4;
            const float q = E[(size_t)d * K_CODES + k];
            const float diff = Af[r * LDA + d] - q;
            sse += diff * diff;
            ob[(size_t)d * HW + r] = q;
        }
    }
    sse_red[tid] = sse;
    __syncthreads();
    for (int s = 128; s > 0; s >>= 1) {
        if (tid < s) sse_red[tid] += sse_red[tid + s];
        __syncthreads();
    }
    if (tid == 0) partials[blockIdx.x] = sse_red[0];
}

// ---------------------------------------------------------------------------
// Finalize: reduce per-block SSE -> both losses; counts -> entropy.
// ---------------------------------------------------------------------------
__global__ void vq_finalize(const float* __restrict__ partials,
                            const unsigned* __restrict__ counts,
                            float* __restrict__ out) {
    __shared__ float red[1024];
    const int t = threadIdx.x;
    float s = 0.0f;
    for (int i = t; i < N_PTS / M_TILE; i += 1024) s += partials[i];
    red[t] = s;
    __syncthreads();
    for (int st = 512; st > 0; st >>= 1) {
        if (t < st) red[t] += red[t + st];
        __syncthreads();
    }
    if (t == 0) {
        const float loss = red[0] / (float)TOTAL_ELEMS;
        out[TOTAL_ELEMS + 0] = loss;   // dictionary_loss
        out[TOTAL_ELEMS + 1] = loss;   // commitment_loss (== fwd)
    }
    __syncthreads();
    const float p = (float)counts[t] / (float)N_PTS;
    red[t] = -p * logf(p + 1e-10f);
    __syncthreads();
    for (int st = 512; st > 0; st >>= 1) {
        if (t < st) red[t] += red[t + st];
        __syncthreads();
    }
    if (t == 0) out[TOTAL_ELEMS + 2] = red[0];
}

// ---------------------------------------------------------------------------
// d_out layout (float): quantized_ste [0,16777216), dict loss, commit loss,
// entropy, indices [16777219, +65536)
// ---------------------------------------------------------------------------
extern "C" void kernel_launch(void* const* d_in, const int* in_sizes, int n_in,
                              void* d_out, int out_size, void* d_ws, size_t ws_size,
                              hipStream_t stream) {
    (void)in_sizes; (void)n_in; (void)out_size; (void)ws_size;
    const float* x = (const float*)d_in[0];
    const float* E = (const float*)d_in[1];
    float* out = (float*)d_out;

    char* ws = (char*)d_ws;
    _Float16* Et      = (_Float16*)ws;                         // 1024*256*2 = 512 KB
    float*    enorm   = (float*)(ws + 524288);                 // 4 KB
    unsigned* counts  = (unsigned*)(ws + 524288 + 4096);       // 4 KB
    float*    parts   = (float*)(ws + 524288 + 8192);          // 4 KB

    vq_prep<<<K_CODES, 256, 0, stream>>>(E, Et, enorm, counts);
    vq_main<<<N_PTS / M_TILE, 256, 0, stream>>>(x, E, Et, enorm,
                                                out, out + TOTAL_ELEMS + 3,
                                                counts, parts);
    vq_finalize<<<1, 1024, 0, stream>>>(parts, counts, out);
}
